// FeatureAugmentationNetwork2_49589692400271
// MI455X (gfx1250) — matrix operand order repaired and checked by reading, not compiled, not run on hardware
//
#include <hip/hip_runtime.h>
#include <hip/hip_bf16.h>

// ---------------------------------------------------------------------------
// FeatureAugmentationNetwork2: fused attention-augmentation for MI455X/gfx1250
//   q = X Wq^T + bq ; k = Mem Wk^T + bk          (WMMA bf16 GEMMs -> bf16)
//   out = 0.5*X + 0.5*softmax(q k^T) Mem         (single-pass flash softmax)
// All matrix math via v_wmma_f32_16x16x32_bf16 (wave32 WMMA).
// Total matrix work = roofline-minimal ~146 GFLOP (no softmax-stat pre-pass).
// ---------------------------------------------------------------------------

typedef __bf16 bf16_t;
typedef __attribute__((ext_vector_type(16))) __bf16 v16bf;
typedef __attribute__((ext_vector_type(8)))  __bf16 v8bf;
typedef __attribute__((ext_vector_type(8)))  float  v8f;

static constexpr int Ndim = 8192;   // rows of features
static constexpr int Mdim = 8192;   // rows of memory_features
static constexpr int Hdim = 512;    // hidden

__device__ __forceinline__ v8f wmma_bf16(v16bf a, v16bf b, v8f c) {
  return __builtin_amdgcn_wmma_f32_16x16x32_bf16(
      /*neg_a=*/false, a, /*neg_b=*/false, b,
      /*c_mod=*/(short)0, c, /*reuse_a=*/false, /*reuse_b=*/false);
}

// A fragment (16x32, M x K), bf16 row-major source, row stride ld elements.
// Lane L: M = L%16 ; elements 0..7 -> K = (L/16)*8 + e ; 8..15 -> K = 16 + (L/16)*8 + e
__device__ __forceinline__ v16bf load_frag_a_bf16(const bf16_t* p, int ld, int lane) {
  const int m  = lane & 15;
  const int kh = lane >> 4;
  const bf16_t* r = p + (size_t)m * ld + kh * 8;
  v8bf lo = *(const v8bf*)(r);
  v8bf hi = *(const v8bf*)(r + 16);
  v16bf a;
#pragma unroll
  for (int e = 0; e < 8; ++e) { a[e] = lo[e]; a[e + 8] = hi[e]; }
  return a;
}

// B fragment (32x16, K x N) where B[k,n] = src[n*ld + k] (K contiguous).
// Lane L: N = L%16 ; elements 0..7 -> K = (L/16)*8 + e ; 8..15 -> K = 16 + (L/16)*8 + e
__device__ __forceinline__ v16bf load_frag_b_bf16(const bf16_t* p, int ld, int lane) {
  const int n  = lane & 15;
  const int kh = lane >> 4;
  const bf16_t* r = p + (size_t)n * ld + kh * 8;
  v8bf lo = *(const v8bf*)(r);
  v8bf hi = *(const v8bf*)(r + 16);
  v16bf b;
#pragma unroll
  for (int e = 0; e < 8; ++e) { b[e] = lo[e]; b[e + 8] = hi[e]; }
  return b;
}

// Same fragments but sourced from f32, converted to bf16 on load (projections).
__device__ __forceinline__ v16bf load_frag_a_f32(const float* p, int ld, int lane) {
  const int m  = lane & 15;
  const int kh = lane >> 4;
  const float* r = p + (size_t)m * ld + kh * 8;
  v16bf a;
#pragma unroll
  for (int e = 0; e < 8; ++e) { a[e] = (bf16_t)r[e]; a[e + 8] = (bf16_t)r[e + 16]; }
  return a;
}
__device__ __forceinline__ v16bf load_frag_b_f32(const float* p, int ld, int lane) {
  const int n  = lane & 15;
  const int kh = lane >> 4;
  const float* r = p + (size_t)n * ld + kh * 8;
  v16bf b;
#pragma unroll
  for (int e = 0; e < 8; ++e) { b[e] = (bf16_t)r[e]; b[e + 8] = (bf16_t)r[e + 16]; }
  return b;
}

// ---------------------------------------------------------------------------
// Pass 0: V^T transpose (Mem f32 [M,H] -> bf16 [H,M]) via LDS tile
// ---------------------------------------------------------------------------
__global__ void fan2_transpose_kernel(const float* __restrict__ src,
                                      bf16_t* __restrict__ dst) {
  __shared__ float tile[16][17];
  const int h0 = blockIdx.x * 16;
  const int j0 = blockIdx.y * 16;
  const int tx = threadIdx.x, ty = threadIdx.y;
  tile[ty][tx] = src[(size_t)(j0 + ty) * Hdim + h0 + tx];
  __syncthreads();
  dst[(size_t)(h0 + ty) * Mdim + j0 + tx] = (bf16_t)tile[tx][ty];
}

// ---------------------------------------------------------------------------
// Pass 1: out_bf16[n,h] = sum_k X[n,k]*W[h,k] + b[h]   (y = x W^T + b)
// block = 128 threads (4 waves); wave w -> h tile blockIdx.x*64 + 16*w
// ---------------------------------------------------------------------------
__global__ __launch_bounds__(128)
void fan2_linear_bf16_kernel(const float* __restrict__ X,
                             const float* __restrict__ W,
                             const float* __restrict__ bias,
                             bf16_t* __restrict__ out) {
  const int lane = threadIdx.x & 31;
  const int wave = threadIdx.x >> 5;
  const int m0 = blockIdx.y * 16;
  const int h0 = blockIdx.x * 64 + wave * 16;

  v8f acc = {};
#pragma unroll
  for (int kc = 0; kc < Hdim; kc += 32) {
    v16bf a = load_frag_a_f32(X + (size_t)m0 * Hdim + kc, Hdim, lane);
    v16bf b = load_frag_b_f32(W + (size_t)h0 * Hdim + kc, Hdim, lane);
    acc = wmma_bf16(a, b, acc);
  }
  // C/D layout: VGPR r -> M = r + 8*(lane/16), N = lane%16
  const int n = h0 + (lane & 15);
  const float bv = bias[n];
  const int mbase = m0 + ((lane >> 4) << 3);
#pragma unroll
  for (int r = 0; r < 8; ++r)
    out[(size_t)(mbase + r) * Hdim + n] = (bf16_t)(acc[r] + bv);
}

// ---------------------------------------------------------------------------
// Pass 2 (fused): out = 0.5*X + 0.5 * softmax(q k^T) Mem, flash-2 style.
// block = 4 waves, 16 query rows, full H=512 output (wave w -> cols 128w..).
// Per 128-wide M chunk:
//   1. each wave -> 2 S tiles (32 WMMAs over K=512)
//   2. wave-local row max (shfl) -> LDS ; barrier
//   3. workgroup-consistent running max m ; rescale f32 accumulators & l by
//      exp(m_old-m_new) ; P = exp(S-m) -> shared LDS tile ; barrier
//   4. all waves consume P as A fragments for P·V^T (32 WMMAs) ; barrier
// Epilogue: reduce row-sum l across lanes (shfl) and waves (LDS), normalize,
// blend with features.
// ---------------------------------------------------------------------------
__global__ __launch_bounds__(128)
void fan2_attn_out_kernel(const float* __restrict__ features,
                          const bf16_t* __restrict__ Q,
                          const bf16_t* __restrict__ Kb,
                          const bf16_t* __restrict__ VT,   // [H][M] bf16
                          float* __restrict__ out) {
  constexpr int LDP = 136;                 // padded bf16 row stride (16B-aligned)
  __shared__ bf16_t P[16 * LDP];
  __shared__ float cmax[4][16];            // per-wave chunk row max
  __shared__ float lsum[4][16];            // per-wave row-sum partials (epilogue)

  const int lane = threadIdx.x & 31;
  const int wave = threadIdx.x >> 5;       // 0..3
  const int r0   = blockIdx.x * 16;
  const int mrow = (lane >> 4) << 3;       // lane's 8 rows are mrow..mrow+7
  const int ncol = lane & 15;

  // hoisted q fragments (full K=512) for this row block
  v16bf qf[16];
#pragma unroll
  for (int kc = 0; kc < 16; ++kc)
    qf[kc] = load_frag_a_bf16(Q + (size_t)r0 * Hdim + kc * 32, Hdim, lane);

  float m[8], l[8];
#pragma unroll
  for (int r = 0; r < 8; ++r) { m[r] = -3.0e38f; l[r] = 0.0f; }

  v8f acc[8];
#pragma unroll
  for (int t = 0; t < 8; ++t) acc[t] = (v8f){};

  for (int jc = 0; jc < Mdim; jc += 128) {
    // ---- 1. S tiles for this wave's 32 columns
    const int jbase = jc + wave * 32;
    v8f s0 = {}, s1 = {};
#pragma unroll
    for (int kc = 0; kc < 16; ++kc) {
      v16bf b0 = load_frag_b_bf16(Kb + (size_t)jbase * Hdim + kc * 32, Hdim, lane);
      v16bf b1 = load_frag_b_bf16(Kb + (size_t)(jbase + 16) * Hdim + kc * 32, Hdim, lane);
      s0 = wmma_bf16(qf[kc], b0, s0);
      s1 = wmma_bf16(qf[kc], b1, s1);
    }
    // ---- 2. wave-local row max over its 32 cols (lanes within same half)
    float wmax[8];
#pragma unroll
    for (int r = 0; r < 8; ++r) {
      float v = fmaxf(s0[r], s1[r]);
#pragma unroll
      for (int off = 1; off < 16; off <<= 1)
        v = fmaxf(v, __shfl_xor(v, off, 32));
      wmax[r] = v;
    }
    if (ncol == 0) {
#pragma unroll
      for (int r = 0; r < 8; ++r) cmax[wave][mrow + r] = wmax[r];
    }
    __syncthreads();
    // ---- 3. running max update, rescale accumulators & l, P -> LDS
    float f[8];
#pragma unroll
    for (int r = 0; r < 8; ++r) {
      float mn = m[r];
      mn = fmaxf(mn, cmax[0][mrow + r]);
      mn = fmaxf(mn, cmax[1][mrow + r]);
      mn = fmaxf(mn, cmax[2][mrow + r]);
      mn = fmaxf(mn, cmax[3][mrow + r]);
      f[r] = __expf(m[r] - mn);
      m[r] = mn;
    }
#pragma unroll
    for (int t = 0; t < 8; ++t)
#pragma unroll
      for (int r = 0; r < 8; ++r) acc[t][r] *= f[r];
    const int col0 = wave * 32 + ncol;
#pragma unroll
    for (int r = 0; r < 8; ++r) {
      const float p0 = __expf(s0[r] - m[r]);
      const float p1 = __expf(s1[r] - m[r]);
      l[r] = l[r] * f[r] + p0 + p1;        // lane-partial row sum (linear in f)
      P[(mrow + r) * LDP + col0]      = (bf16_t)p0;
      P[(mrow + r) * LDP + col0 + 16] = (bf16_t)p1;
    }
    __syncthreads();
    // ---- 4. P (16x128) @ V^T chunk -> this wave's 16 x 128 output slab
#pragma unroll
    for (int kk = 0; kk < 4; ++kk) {
      const bf16_t* pa = &P[ncol * LDP + kk * 32 + (lane >> 4) * 8];
      v8bf lo = *(const v8bf*)(pa);
      v8bf hi = *(const v8bf*)(pa + 16);
      v16bf afrag;
#pragma unroll
      for (int e = 0; e < 8; ++e) { afrag[e] = lo[e]; afrag[e + 8] = hi[e]; }
      const int kglob = jc + kk * 32;
#pragma unroll
      for (int t = 0; t < 8; ++t) {
        const int h0 = wave * 128 + t * 16;
        v16bf bfrag = load_frag_b_bf16(VT + (size_t)h0 * Mdim + kglob, Mdim, lane);
        acc[t] = wmma_bf16(afrag, bfrag, acc[t]);
      }
    }
    __syncthreads();
  }

  // ---- epilogue: finish l reduction (16 lanes, then 4 waves via LDS)
#pragma unroll
  for (int r = 0; r < 8; ++r) {
#pragma unroll
    for (int off = 1; off < 16; off <<= 1)
      l[r] += __shfl_xor(l[r], off, 32);
  }
  if (ncol == 0) {
#pragma unroll
    for (int r = 0; r < 8; ++r) lsum[wave][mrow + r] = l[r];
  }
  __syncthreads();
  float invl[8];
#pragma unroll
  for (int r = 0; r < 8; ++r) {
    const float lt = lsum[0][mrow + r] + lsum[1][mrow + r] +
                     lsum[2][mrow + r] + lsum[3][mrow + r];
    invl[r] = 1.0f / lt;
  }

  // ---- normalize by l, blend with features (c = 0.5)
#pragma unroll
  for (int t = 0; t < 8; ++t) {
    const int h = wave * 128 + t * 16 + ncol;
#pragma unroll
    for (int r = 0; r < 8; ++r) {
      const size_t i = (size_t)(r0 + mrow + r);
      out[i * Hdim + h] = 0.5f * features[i * Hdim + h] + 0.5f * (acc[t][r] * invl[r]);
    }
  }
}

// ---------------------------------------------------------------------------
// Launch
// ---------------------------------------------------------------------------
extern "C" void kernel_launch(void* const* d_in, const int* in_sizes, int n_in,
                              void* d_out, int out_size, void* d_ws, size_t ws_size,
                              hipStream_t stream) {
  (void)in_sizes; (void)n_in; (void)out_size; (void)ws_size;
  const float* features = (const float*)d_in[0];   // [N,H]
  const float* memf     = (const float*)d_in[1];   // [M,H]
  const float* Wq       = (const float*)d_in[2];   // [H,H]
  const float* bq       = (const float*)d_in[3];   // [H]
  const float* Wk       = (const float*)d_in[4];   // [H,H]
  const float* bk       = (const float*)d_in[5];   // [H]
  float* out = (float*)d_out;

  // workspace layout (bytes): q 8MB | k 8MB | V^T 8MB   (24 MB total)
  char* ws = (char*)d_ws;
  bf16_t* qb = (bf16_t*)(ws);
  bf16_t* kb = (bf16_t*)(ws + (size_t)8  * 1024 * 1024);
  bf16_t* vT = (bf16_t*)(ws + (size_t)16 * 1024 * 1024);

  // Pass 0: Mem -> bf16 V^T [H][M]
  fan2_transpose_kernel<<<dim3(Hdim / 16, Mdim / 16), dim3(16, 16), 0, stream>>>(memf, vT);
  // Pass 1: q, k projections (bf16)
  fan2_linear_bf16_kernel<<<dim3(Hdim / 64, Ndim / 16), 128, 0, stream>>>(features, Wq, bq, qb);
  fan2_linear_bf16_kernel<<<dim3(Hdim / 64, Mdim / 16), 128, 0, stream>>>(memf,     Wk, bk, kb);
  // Pass 2: fused flash attention + blend
  fan2_attn_out_kernel<<<Ndim / 16, 128, 0, stream>>>(features, qb, kb, vT, out);
}
